// NestedEventMamba_67740224192692
// MI455X (gfx1250) — matrix operand order repaired and checked by reading, not compiled, not run on hardware
//
#include <hip/hip_runtime.h>

typedef __attribute__((ext_vector_type(16))) _Float16 v16h;
typedef __attribute__((ext_vector_type(8)))  _Float16 v8h;
typedef __attribute__((ext_vector_type(8)))  float    v8f;

#define DEV __device__ __forceinline__

DEV float siluf(float x) { return x / (1.f + __expf(-x)); }
DEV float geluf(float x) { return 0.5f * x * (1.f + erff(x * 0.70710678118f)); }

// ---------------------------------------------------------------------------
// Generic WMMA GEMM:  C[M,N] = A[M,K] * W[N,K]^T   (y = x @ W^T convention)
//   A  : LDS, f16 row-major, leading dim lda (multiple of 32 halfs)
//   W  : global, f16 row-major [N, ldb], zero-padded; L2-resident weights.
//   epi: called once per output element with (m, n, value)
// K is a compile-time constant at every call site.
//   K <= 256: column-strip mapping — each wave owns N-strips, loads the B
//             (weight) fragments for the whole K once into VGPRs, then streams
//             A tiles from LDS over all M tiles (B traffic cut by M/16).
//   K >  256: streaming fallback (avoids 128-VGPR B arrays next to scan state).
// Fragments use contiguous k (lane half h holds k = 16h..16h+15) for both A and B.
// ---------------------------------------------------------------------------
template <int K, class Epi>
DEV void gemm_wmma(const _Float16* __restrict__ A, int lda,
                   const _Float16* __restrict__ W, int ldb,
                   int M, int N, Epi epi)
{
    constexpr int KF = K / 32;
    const int wave = (int)(threadIdx.x >> 5);
    const int lane = (int)(threadIdx.x & 31);
    const int half = lane >> 4;
    const int r    = lane & 15;
    const int mt_n = M >> 4, nt_n = N >> 4;

    if constexpr (KF <= 8) {
        for (int nt = wave; nt < nt_n; nt += 8) {
            const _Float16* brow = W + (size_t)(nt * 16 + r) * ldb;
            v16h bfr[KF];
#pragma unroll
            for (int kf = 0; kf < KF; ++kf)
                bfr[kf] = *(const v16h*)(brow + kf * 32 + half * 16);
            for (int mt = 0; mt < mt_n; ++mt) {
                v8f c = {};
                const _Float16* arow = A + (size_t)(mt * 16 + r) * lda;
#pragma unroll
                for (int kf = 0; kf < KF; ++kf) {
                    v16h av = *(const v16h*)(arow + kf * 32 + half * 16);
                    c = __builtin_amdgcn_wmma_f32_16x16x32_f16(
                            false, av, false, bfr[kf], (short)0, c, false, false);
                }
#pragma unroll
                for (int i = 0; i < 8; ++i)
                    epi(mt * 16 + half * 8 + i, nt * 16 + r, c[i]);
            }
        }
    } else {
        const int t_tot = mt_n * nt_n;
        for (int tile = wave; tile < t_tot; tile += 8) {
            const int mt = tile / nt_n, nt = tile - mt * nt_n;
            v8f c = {};
            const _Float16* arow = A + (size_t)(mt * 16 + r) * lda;
            const _Float16* brow = W + (size_t)(nt * 16 + r) * ldb;
#pragma unroll
            for (int kf = 0; kf < KF; ++kf) {
                v16h av = *(const v16h*)(arow + kf * 32 + half * 16);
                v16h bv = *(const v16h*)(brow + kf * 32 + half * 16);
                c = __builtin_amdgcn_wmma_f32_16x16x32_f16(
                        false, av, false, bv, (short)0, c, false, false);
            }
#pragma unroll
            for (int i = 0; i < 8; ++i)
                epi(mt * 16 + half * 8 + i, nt * 16 + r, c[i]);
        }
    }
}

// ---------------------------------------------------------------------------
// Per-block weights (f32 small vectors stay in global; GEMM matrices are
// pre-converted to padded f16 in the workspace).
// ---------------------------------------------------------------------------
struct BlockW {
    const float *norm_w, *lc_w, *lc_b, *lnc_w, *lnc_b;
    const float *conv_w, *conv_b, *dt_bias, *A_log, *Dp;
    const float *pc_w, *pc_b, *lnp_w, *lnp_b, *b1v, *b2v;
    const _Float16 *in_proj;   // [2*DI, D]
    const _Float16 *x_proj;    // [NXP,  DI]  (N padded to 16)
    const _Float16 *dt_proj;   // [DI,   32]  (K padded 8/16 -> 32)
    const _Float16 *out_proj;  // [D,    DI]
    const _Float16 *w1;        // [4D,   D]
    const _Float16 *w2;        // [D,    4D]
};

// ---------------------------------------------------------------------------
// Fused conv-mamba block:  x += SiLU(LN(pwconv(Mamba(xc+xn)))) + MLP(xn)
// Sequence chunked (CHUNK) so all activations live in LDS; SSM state h[d][n]
// and A[d][n] live in VGPRs of the thread owning channel d (CPT = DI/256).
// ---------------------------------------------------------------------------
template <int L, int D, int DI, int DTR, int CHUNK>
DEV void mamba_block(_Float16* x, _Float16* xn, _Float16* b, _Float16* xs,
                     _Float16* cp, _Float16* cx, _Float16* cg, _Float16* cd,
                     _Float16* dtin, float* bc, _Float16* tail,
                     const BlockW& w, int lck, int lcpad)
{
    constexpr int CPT = DI / 256;
    constexpr int NX  = DTR + 64;
    constexpr int NXP = (NX + 15) & ~15;
    const int tid = (int)threadIdx.x;

    // Warm L2 with this block's weights (global_prefetch_b8).
    if (tid < 8) {
        const _Float16* wp[8] = { w.in_proj, w.x_proj, w.dt_proj, w.out_proj,
                                  w.w1, w.w2, w.in_proj + (size_t)DI * D,
                                  w.w2 + 2 * D };
        __builtin_prefetch(wp[tid], 0, 1);
    }

    float Areg[CPT][32], hreg[CPT][32], Dreg[CPT];
#pragma unroll
    for (int j = 0; j < CPT; ++j) {
        const int d = tid + j * 256;
        Dreg[j] = w.Dp[d];
#pragma unroll
        for (int n = 0; n < 32; ++n) {
            Areg[j][n] = -__expf(w.A_log[d * 32 + n]);
            hreg[j][n] = 0.f;
        }
    }

    // -------- RMSNorm: xn = rmsnorm(x) --------
    if (tid < L) {
        float ss = 0.f;
        for (int dd = 0; dd < D; ++dd) { float v = (float)x[tid * D + dd]; ss += v * v; }
        const float sc = rsqrtf(ss / (float)D + 1e-6f);
        for (int dd = 0; dd < D; ++dd)
            xn[tid * D + dd] = (_Float16)((float)x[tid * D + dd] * sc * w.norm_w[dd]);
    }
    __syncthreads();

    // -------- MLP: x += gelu(xn@w1^T+b1)@w2^T+b2  (K chunked by D, t buf=xs)
    for (int c = 0; c < 4; ++c) {
        const int cs = c * D;
        gemm_wmma<D>(xn, D, w.w1 + (size_t)cs * D, D, L, D,
                     [&](int m, int n, float v) {
                         xs[m * D + n] = (_Float16)geluf(v + w.b1v[cs + n]);
                     });
        __syncthreads();
        gemm_wmma<D>(xs, D, w.w2 + cs, 4 * D, L, D,
                     [&](int m, int n, float v) {
                         x[m * D + n] = (_Float16)((float)x[m * D + n] + v);
                     });
        __syncthreads();
    }
    for (int i = tid; i < L * D; i += 256)
        x[i] = (_Float16)((float)x[i] + w.b2v[i % D]);
    __syncthreads();

    // -------- lc depthwise conv (k=lck,pad=lcpad) + LN + SiLU; b = xc + xn --
    for (int i = tid; i < L * D; i += 256) {
        const int t = i / D, dd = i % D;
        float acc = w.lc_b[dd];
        for (int j = 0; j < lck; ++j) {
            const int tt = t - lcpad + j;
            if (tt >= 0 && tt < L) acc += (float)xn[tt * D + dd] * w.lc_w[dd * lck + j];
        }
        xs[i] = (_Float16)acc;
    }
    __syncthreads();
    if (tid < L) {
        float m = 0.f;
        for (int dd = 0; dd < D; ++dd) m += (float)xs[tid * D + dd];
        m /= (float)D;
        float v = 0.f;
        for (int dd = 0; dd < D; ++dd) { float u = (float)xs[tid * D + dd] - m; v += u * u; }
        const float rs = rsqrtf(v / (float)D + 1e-5f);
        for (int dd = 0; dd < D; ++dd) {
            const float u = ((float)xs[tid * D + dd] - m) * rs * w.lnc_w[dd] + w.lnc_b[dd];
            b[tid * D + dd] = (_Float16)(siluf(u) + (float)xn[tid * D + dd]);
        }
    }
    __syncthreads();

    // -------- Mamba, chunked over the sequence --------
    for (int c0 = 0; c0 < L; c0 += CHUNK) {
        // in_proj: xi_pre -> cp ; gate g = silu(z) -> cg
        gemm_wmma<D>(b + (size_t)c0 * D, D, w.in_proj, D, CHUNK, DI,
                     [&](int m, int n, float v) { cp[m * DI + n] = (_Float16)v; });
        gemm_wmma<D>(b + (size_t)c0 * D, D, w.in_proj + (size_t)DI * D, D, CHUNK, DI,
                     [&](int m, int n, float v) { cg[m * DI + n] = (_Float16)siluf(v); });
        __syncthreads();

        // causal depthwise conv width 4 + SiLU -> cx (tail = prev chunk rows)
        for (int i = tid; i < CHUNK * DI; i += 256) {
            const int tl = i / DI, dd = i % DI;
            float acc = w.conv_b[dd];
#pragma unroll
            for (int j = 0; j < 4; ++j) {
                const int ts = tl - 3 + j;
                float xv = 0.f;
                if (ts >= 0)          xv = (float)cp[ts * DI + dd];
                else if (c0 > 0)      xv = (float)tail[(ts + 3) * DI + dd];
                acc += xv * w.conv_w[dd * 4 + j];
            }
            cx[i] = (_Float16)siluf(acc);
        }
        __syncthreads();
        for (int i = tid; i < 3 * DI; i += 256) tail[i] = cp[(CHUNK - 3) * DI + i];
        for (int i = tid; i < CHUNK * 32; i += 256) dtin[i] = (_Float16)0.f;
        __syncthreads();

        // x_proj: split into dt-input (padded K=32) and B/C (f32)
        gemm_wmma<DI>(cx, DI, w.x_proj, DI, CHUNK, NXP,
                      [&](int m, int n, float v) {
                          if (n < DTR)      dtin[m * 32 + n] = (_Float16)v;
                          else if (n < NX)  bc[m * 64 + (n - DTR)] = v;
                      });
        __syncthreads();

        // dt = softplus(dtin @ dt_proj^T + dt_bias)
        gemm_wmma<32>(dtin, 32, w.dt_proj, 32, CHUNK, DI,
                      [&](int m, int n, float v) {
                          v += w.dt_bias[n];
                          cd[m * DI + n] = (_Float16)((v > 20.f) ? v : log1pf(__expf(v)));
                      });
        __syncthreads();

        // selective scan: thread owns channel(s) d, state in VGPRs.
        for (int tl = 0; tl < CHUNK; ++tl) {
#pragma unroll
            for (int j = 0; j < CPT; ++j) {
                const int d = tid + j * 256;
                const float dtv = (float)cd[tl * DI + d];
                const float xiv = (float)cx[tl * DI + d];
                const float dx  = dtv * xiv;
                float acc = Dreg[j] * xiv;
#pragma unroll
                for (int n = 0; n < 32; ++n) {
                    const float h = __expf(dtv * Areg[j][n]) * hreg[j][n]
                                  + dx * bc[tl * 64 + n];
                    hreg[j][n] = h;
                    acc += h * bc[tl * 64 + 32 + n];
                }
                const float g = (float)cg[tl * DI + d];
                cg[tl * DI + d] = (_Float16)(acc * g);   // gated y, in place
            }
        }
        __syncthreads();

        // out_proj -> xs rows of this chunk
        gemm_wmma<DI>(cg, DI, w.out_proj, DI, CHUNK, D,
                      [&](int m, int n, float v) { xs[(c0 + m) * D + n] = (_Float16)v; });
        __syncthreads();
    }

    // -------- pointwise conv k=3 + LN + SiLU; x += xm --------
    for (int i = tid; i < L * D; i += 256) {
        const int t = i / D, dd = i % D;
        float acc = w.pc_b[dd];
#pragma unroll
        for (int j = 0; j < 3; ++j) {
            const int tt = t - 1 + j;
            if (tt >= 0 && tt < L) acc += (float)xs[tt * D + dd] * w.pc_w[dd * 3 + j];
        }
        b[i] = (_Float16)acc;
    }
    __syncthreads();
    if (tid < L) {
        float m = 0.f;
        for (int dd = 0; dd < D; ++dd) m += (float)b[tid * D + dd];
        m /= (float)D;
        float v = 0.f;
        for (int dd = 0; dd < D; ++dd) { float u = (float)b[tid * D + dd] - m; v += u * u; }
        const float rs = rsqrtf(v / (float)D + 1e-5f);
        for (int dd = 0; dd < D; ++dd) {
            const float u = ((float)b[tid * D + dd] - m) * rs * w.lnp_w[dd] + w.lnp_b[dd];
            x[tid * D + dd] = (_Float16)((float)x[tid * D + dd] + siluf(u));
        }
    }
    __syncthreads();
}

// ---------------------------------------------------------------------------
// Kernel 1: one workgroup per window (128 WGs, 256 threads = 8 waves).
// events -> encoder MLP -> 2 conv-mamba blocks -> (mean+max)/2 pooling.
// LDS arena ~272KB (f16 activations), 1 WG/WGP.
// ---------------------------------------------------------------------------
__global__ void __launch_bounds__(256)
intra_kernel(const float* __restrict__ events,
             const _Float16* __restrict__ ia_w1h, const float* __restrict__ ia_b1,
             const _Float16* __restrict__ ia_w2h, const float* __restrict__ ia_b2,
             BlockW b0, BlockW b1, float* __restrict__ wv)
{
    extern __shared__ char smraw[];
    _Float16* sm = (_Float16*)smraw;
    _Float16* x    = sm;               // [256,128]
    _Float16* xn   = sm + 32768;       // [256,128]; chunk bufs overlay it
    _Float16* b    = sm + 65536;       // [256,128]
    _Float16* xs   = sm + 98304;       // [256,128]
    _Float16* cp   = xn;               // [32,256] each
    _Float16* cx   = xn + 8192;
    _Float16* cg   = xn + 16384;
    _Float16* cd   = xn + 24576;
    _Float16* extra = sm + 131072;     // 8192 halfs
    _Float16* dtin = extra;            // [32,32]
    float*    bc   = (float*)(extra + 1024); // [32,64] f32
    _Float16* tail = extra + 1024 + 4096;    // [3,256]

    const int tid = (int)threadIdx.x;
    const int win = (int)blockIdx.x;   // 0..127  (= batch*32 + window)

    // events row-load, F=130 zero-padded to 160 (ev overlays xn + head of b)
    _Float16* ev = xn;
    for (int i = tid; i < 256 * 160; i += 256) {
        const int t = i / 160, f = i % 160;
        ev[i] = (f < 130) ? (_Float16)events[((size_t)win * 256 + t) * 130 + f]
                          : (_Float16)0.f;
    }
    __syncthreads();

    gemm_wmma<160>(ev, 160, ia_w1h, 160, 256, 128,
                   [&](int m, int n, float v) { xs[m * 128 + n] = (_Float16)geluf(v + ia_b1[n]); });
    __syncthreads();
    gemm_wmma<128>(xs, 128, ia_w2h, 128, 256, 128,
                   [&](int m, int n, float v) { x[m * 128 + n] = (_Float16)(v + ia_b2[n]); });
    __syncthreads();

    mamba_block<256, 128, 256, 8, 32>(x, xn, b, xs, cp, cx, cg, cd, dtin, bc, tail, b0, 11, 5);
    mamba_block<256, 128, 256, 8, 32>(x, xn, b, xs, cp, cx, cg, cd, dtin, bc, tail, b1, 11, 5);

    // window pooling: (mean + max) / 2 over the 256 tokens
    if (tid < 128) {
        float s = 0.f, mx = -3.0e38f;
        for (int t = 0; t < 256; ++t) {
            const float v = (float)x[t * 128 + tid];
            s += v; mx = fmaxf(mx, v);
        }
        wv[(size_t)win * 128 + tid] = 0.5f * (s / 256.f + mx);
    }
}

// ---------------------------------------------------------------------------
// Kernel 2: one workgroup per batch element. stage1 (L=32,D=128) ->
// stride-2 downsample + LN -> stage2 (L=16,D=256) -> attn pool -> head.
// ---------------------------------------------------------------------------
__global__ void __launch_bounds__(256)
seq_kernel(const float* __restrict__ wv, BlockW s1, BlockW s2,
           const float* __restrict__ ds_w,   const float* __restrict__ ds_b,
           const float* __restrict__ ds_ln_w,const float* __restrict__ ds_ln_b,
           const float* __restrict__ pool_w, const float* __restrict__ pool_b,
           const float* __restrict__ h_ln_w, const float* __restrict__ h_ln_b,
           const float* __restrict__ h_w1,   const float* __restrict__ h_b1,
           const float* __restrict__ h_w2,   const float* __restrict__ h_b2,
           float* __restrict__ out)
{
    extern __shared__ char smraw[];
    _Float16* sm = (_Float16*)smraw;
    _Float16* x    = sm;               // max [32,128] / [16,256]
    _Float16* xn   = sm + 4096;
    _Float16* b    = sm + 8192;
    _Float16* xs   = sm + 12288;
    _Float16* cp   = sm + 16384;       // 8192 halfs each
    _Float16* cx   = sm + 24576;
    _Float16* cg   = sm + 32768;
    _Float16* cd   = sm + 40960;
    _Float16* dtin = sm + 49152;       // 1024
    float*    bc   = (float*)(sm + 50176); // 2048 floats
    _Float16* tail = sm + 54272;       // 1536

    const int tid = (int)threadIdx.x;
    const int bat = (int)blockIdx.x;

    for (int i = tid; i < 32 * 128; i += 256)
        x[i] = (_Float16)wv[(size_t)bat * 4096 + i];
    __syncthreads();

    mamba_block<32, 128, 256, 8, 32>(x, xn, b, xs, cp, cx, cg, cd, dtin, bc, tail, s1, 3, 1);

    // downsample: stride-2 conv [32,128] -> [16,256], then LN -> x
    for (int i = tid; i < 16 * 256; i += 256) {
        const int t2 = i >> 8, o = i & 255;
        float acc = ds_b[o];
        for (int ii = 0; ii < 128; ++ii)
            acc += (float)x[(2 * t2) * 128 + ii]     * ds_w[(o * 128 + ii) * 2 + 0]
                 + (float)x[(2 * t2 + 1) * 128 + ii] * ds_w[(o * 128 + ii) * 2 + 1];
        b[i] = (_Float16)acc;
    }
    __syncthreads();
    if (tid < 16) {
        float m = 0.f;
        for (int o = 0; o < 256; ++o) m += (float)b[tid * 256 + o];
        m /= 256.f;
        float v = 0.f;
        for (int o = 0; o < 256; ++o) { float u = (float)b[tid * 256 + o] - m; v += u * u; }
        const float rs = rsqrtf(v / 256.f + 1e-5f);
        for (int o = 0; o < 256; ++o)
            x[tid * 256 + o] = (_Float16)(((float)b[tid * 256 + o] - m) * rs * ds_ln_w[o] + ds_ln_b[o]);
    }
    __syncthreads();

    mamba_block<16, 256, 512, 16, 16>(x, xn, b, xs, cp, cx, cg, cd, dtin, bc, tail, s2, 3, 1);

    // attention pooling + head (bc float area reused as scratch)
    float* fs = bc;   // [0..15] scores, [16..31] attn, [64..319] pooled, [384..511] hidden
    if (tid < 16) {
        float a = pool_b[0];
        for (int dd = 0; dd < 256; ++dd) a += (float)x[tid * 256 + dd] * pool_w[dd];
        fs[tid] = a;
    }
    __syncthreads();
    if (tid == 0) {
        float mx = -3.0e38f;
        for (int t = 0; t < 16; ++t) mx = fmaxf(mx, fs[t]);
        float se = 0.f;
        for (int t = 0; t < 16; ++t) { float e = __expf(fs[t] - mx); fs[16 + t] = e; se += e; }
        for (int t = 0; t < 16; ++t) fs[16 + t] /= se;
    }
    __syncthreads();
    {
        float p = 0.f;
        for (int t = 0; t < 16; ++t) p += fs[16 + t] * (float)x[t * 256 + tid];
        fs[64 + tid] = p;
    }
    __syncthreads();
    if (tid == 0) {
        float m = 0.f;
        for (int dd = 0; dd < 256; ++dd) m += fs[64 + dd];
        m /= 256.f;
        float v = 0.f;
        for (int dd = 0; dd < 256; ++dd) { float u = fs[64 + dd] - m; v += u * u; }
        const float rs = rsqrtf(v / 256.f + 1e-5f);
        for (int dd = 0; dd < 256; ++dd)
            fs[64 + dd] = (fs[64 + dd] - m) * rs * h_ln_w[dd] + h_ln_b[dd];
    }
    __syncthreads();
    if (tid < 128) {
        float a = h_b1[tid];
        for (int dd = 0; dd < 256; ++dd) a += fs[64 + dd] * h_w1[tid * 256 + dd];
        fs[384 + tid] = geluf(a);
    }
    __syncthreads();
    if (tid < 11) {
        float a = h_b2[tid];
        for (int dd = 0; dd < 128; ++dd) a += fs[384 + dd] * h_w2[tid * 128 + dd];
        out[bat * 11 + tid] = a;
    }
}

// ---------------------------------------------------------------------------
// Weight prep: f32 [O,I] -> f16 [OP,IP] zero-padded (K to mult-32, N to 16).
// ---------------------------------------------------------------------------
__global__ void __launch_bounds__(256)
cvt_pad_kernel(const float* __restrict__ src, _Float16* __restrict__ dst,
               int O, int I, int OP, int IP)
{
    const int total = OP * IP;
    for (int i = (int)(blockIdx.x * 256 + threadIdx.x); i < total; i += (int)(gridDim.x * 256)) {
        const int o = i / IP, c = i % IP;
        dst[i] = (o < O && c < I) ? (_Float16)src[(size_t)o * I + c] : (_Float16)0.f;
    }
}

// ---------------------------------------------------------------------------
// Host launcher. Input order (flattened setup_inputs dict):
//  0 events; 1-4 ia_{w1,b1,w2,b2}; 5..26 intra[0]; 27..48 intra[1];
//  49..70 stage1[0]; 71-74 ds_{w,b,ln_w,ln_b}; 75..96 stage2[0];
//  97-98 pool_{w,b}; 99-104 h_{ln_w,ln_b,w1,b1,w2,b2}.
// ---------------------------------------------------------------------------
extern "C" void kernel_launch(void* const* d_in, const int* in_sizes, int n_in,
                              void* d_out, int out_size, void* d_ws, size_t ws_size,
                              hipStream_t stream)
{
    auto F32 = [&](int i) { return (const float*)d_in[i]; };

    float*     wv  = (float*)d_ws;                        // [128,128] window vecs
    _Float16*  wsh = (_Float16*)((char*)d_ws + 65536);    // f16 weight pool
    size_t     off = 0;

    auto cvt = [&](const float* src, int O, int I, int OP, int IP) -> const _Float16* {
        _Float16* dst = wsh + off;
        off += (size_t)OP * IP;
        const int total = OP * IP;
        int blocks = (total + 255) / 256;
        if (blocks > 1024) blocks = 1024;
        cvt_pad_kernel<<<dim3(blocks), dim3(256), 0, stream>>>(src, dst, O, I, OP, IP);
        return dst;
    };

    const _Float16* ia_w1h = cvt(F32(1), 128, 130, 128, 160);
    const _Float16* ia_w2h = cvt(F32(3), 128, 128, 128, 128);

    auto mkblock = [&](int base, int D, int DI, int DTR) -> BlockW {
        BlockW w;
        w.norm_w = F32(base + 0);  w.lc_w  = F32(base + 1);  w.lc_b  = F32(base + 2);
        w.lnc_w  = F32(base + 3);  w.lnc_b = F32(base + 4);
        w.in_proj = cvt(F32(base + 5), 2 * DI, D, 2 * DI, D);
        w.conv_w = F32(base + 6);  w.conv_b = F32(base + 7);
        const int NX = DTR + 64, NXP = (NX + 15) & ~15;
        w.x_proj  = cvt(F32(base + 8), NX, DI, NXP, DI);
        w.dt_proj = cvt(F32(base + 9), DI, DTR, DI, 32);
        w.dt_bias = F32(base + 10); w.A_log = F32(base + 11); w.Dp = F32(base + 12);
        w.out_proj = cvt(F32(base + 13), D, DI, D, DI);
        w.pc_w = F32(base + 14); w.pc_b = F32(base + 15);
        w.lnp_w = F32(base + 16); w.lnp_b = F32(base + 17);
        w.w1  = cvt(F32(base + 18), 4 * D, D, 4 * D, D);
        w.b1v = F32(base + 19);
        w.w2  = cvt(F32(base + 20), D, 4 * D, D, 4 * D);
        w.b2v = F32(base + 21);
        return w;
    };

    BlockW bi0 = mkblock(5,  128, 256, 8);
    BlockW bi1 = mkblock(27, 128, 256, 8);
    BlockW bs1 = mkblock(49, 128, 256, 8);
    BlockW bs2 = mkblock(75, 256, 512, 16);

    const size_t sm1 = (size_t)139264 * 2;   // 272KB LDS arena (<= 320KB/WG)
    const size_t sm2 = (size_t)55808 * 2;    // 109KB

    intra_kernel<<<dim3(128), dim3(256), sm1, stream>>>(
        F32(0), ia_w1h, F32(2), ia_w2h, F32(4), bi0, bi1, wv);

    seq_kernel<<<dim3(4), dim3(256), sm2, stream>>>(
        wv, bs1, bs2,
        F32(71), F32(72), F32(73), F32(74),
        F32(97), F32(98), F32(99), F32(100),
        F32(101), F32(102), F32(103), F32(104),
        (float*)d_out);

    (void)in_sizes; (void)n_in; (void)out_size; (void)ws_size;
}
